// RNN_8821862826731
// MI455X (gfx1250) — matrix operand order repaired and checked by reading, not compiled, not run on hardware
//
#include <hip/hip_runtime.h>

// ---------------------------------------------------------------------------
// Types for CDNA5 WMMA (wave32): v_wmma_f32_16x16x32_bf16
// ---------------------------------------------------------------------------
typedef __bf16 bf16;
typedef __attribute__((ext_vector_type(16))) __bf16 v16bf;
typedef __attribute__((ext_vector_type(8)))  __bf16 v8bf;
typedef __attribute__((ext_vector_type(8)))  float  v8f;

#define Bc 64
#define Sc 512
#define Ic 512
#define Hc 512
#define Lc 2

__device__ __forceinline__ bf16 f2bf(float f) {
  unsigned u = __builtin_bit_cast(unsigned, f);
  unsigned r = u + 0x7FFFu + ((u >> 16) & 1u);   // round-to-nearest-even
  unsigned short h = (unsigned short)(r >> 16);
  return __builtin_bit_cast(bf16, h);
}

// CDNA5 async global->LDS copy (ASYNCcnt path, CDNA5 ISA).
// vdst = LDS byte address (low 32 bits of a generic shared pointer),
// vaddr = 64-bit global address, GV mode.
__device__ __forceinline__ void async_g2lds_b128(const float* g, const float* lds_ptr) {
  unsigned lds = (unsigned)(size_t)lds_ptr;
  asm volatile("global_load_async_to_lds_b128 %0, %1, off"
               :: "v"(lds), "v"(g) : "memory");
}
__device__ __forceinline__ void wait_async0() {
  asm volatile("s_wait_asynccnt 0" ::: "memory");
}

// ---------------------------------------------------------------------------
// f32 -> bf16 conversion (grid-stride)
// ---------------------------------------------------------------------------
__global__ void cvt_f32_bf16(const float* __restrict__ src,
                             bf16* __restrict__ dst, int n) {
  int i = blockIdx.x * blockDim.x + threadIdx.x;
  int stride = gridDim.x * blockDim.x;
  for (; i < n; i += stride) dst[i] = f2bf(src[i]);
}

// ---------------------------------------------------------------------------
// Kernel 1: xa[B*S, H] = x[B*S, I] * W_ih^T + b_ih   (bf16 WMMA, f32 acc)
// Register-blocked: each wave owns a 64x32 output tile (4 M-tiles x 2 N-tiles)
// -> 8 WMMAs per K-chunk fed by 4 A-fragments + 2 B-fragments (A reused over
// N, B reused over M).
// ---------------------------------------------------------------------------
__global__ void __launch_bounds__(256)
gemm_xa_kernel(const bf16*  __restrict__ xb,   // [B*S, I] bf16
               const bf16*  __restrict__ wib,  // [H, I]   bf16 (row n = B column n)
               const float* __restrict__ bih,  // [H]
               float*       __restrict__ xa)   // [B*S, H] f32
{
  const int lane = threadIdx.x & 31;
  const int wave = threadIdx.x >> 5;
  const int tile = blockIdx.x * 8 + wave;      // 8192 wave-supertiles
  const int NS   = Hc / 32;                    // 16 N-supertiles (32 cols each)
  const int ms   = tile / NS;                  // 0..511 (64 rows each)
  const int ns   = tile - ms * NS;             // 0..15

  const int half = lane >> 4;                  // 0..1
  const int l15  = lane & 15;

  const bf16* aptr[4];
  #pragma unroll
  for (int mi = 0; mi < 4; ++mi)
    aptr[mi] = xb + (size_t)(ms * 64 + mi * 16 + l15) * Ic;
  const bf16* bptr[2];
  #pragma unroll
  for (int ni = 0; ni < 2; ++ni)
    bptr[ni] = wib + (size_t)(ns * 32 + ni * 16 + l15) * Ic;

  v8f acc[4][2] = {};
  for (int kc = 0; kc < Ic / 32; ++kc) {
    const int k0 = kc * 32 + half * 16;        // A: two 8-elem chunks
    const int kb = kc * 32 + half * 16;        // B: one 16-elem chunk
    v16bf b[2];
    #pragma unroll
    for (int ni = 0; ni < 2; ++ni)
      b[ni] = *(const v16bf*)(bptr[ni] + kb);
    #pragma unroll
    for (int mi = 0; mi < 4; ++mi) {
      union { v16bf v; v8bf h[2]; } a;
      a.h[0] = *(const v8bf*)(aptr[mi] + k0);
      a.h[1] = *(const v8bf*)(aptr[mi] + k0 + 8);
      #pragma unroll
      for (int ni = 0; ni < 2; ++ni)
        acc[mi][ni] = __builtin_amdgcn_wmma_f32_16x16x32_bf16(
            false, a.v, false, b[ni], (short)0, acc[mi][ni], false, false);
    }
  }

  #pragma unroll
  for (int ni = 0; ni < 2; ++ni) {
    const int col  = ns * 32 + ni * 16 + l15;
    const float bias = bih[col];
    #pragma unroll
    for (int mi = 0; mi < 4; ++mi) {
      float* orow = xa + (size_t)(ms * 64 + mi * 16 + half * 8) * Hc + col;
      #pragma unroll
      for (int r = 0; r < 8; ++r)
        orow[(size_t)r * Hc] = acc[mi][ni][r] + bias;
    }
  }
}

// ---------------------------------------------------------------------------
// Kernel 2: serial scan over S=512 steps.
// 8 blocks of one 16-row M-tile each; live h tile in LDS (16 KB) for all
// steps; W_hh bf16 (512 KB) L2-resident. The 16 KB xa slice for step t+1 is
// prefetched into a double-buffered LDS region via GLOBAL_LOAD_ASYNC_TO_LDS
// while the 64 WMMAs/wave of step t execute; s_wait_asynccnt 0 + barrier
// publishes it. y stores are non-temporal (write-once stream, 134 MB).
// ---------------------------------------------------------------------------
__global__ void __launch_bounds__(256)
rnn_scan_kernel(const float* __restrict__ xa,    // [B*S, H] f32
                const bf16*  __restrict__ whb,   // [H, H] bf16 (row g = B column g)
                const float* __restrict__ bhh,   // [H]
                float*       __restrict__ y,     // [B, S, L, H] f32
                float*       __restrict__ hlast) // [B, L, H]   f32
{
  __shared__ bf16  hbuf[16 * Hc];                // 16 KB live state
  __shared__ float xabuf[2][8 * Hc];             // 2 x 16 KB xa double-buffer

  const int tid  = threadIdx.x;
  const int lane = tid & 31;
  const int wave = tid >> 5;
  const int half = lane >> 4;
  const int l15  = lane & 15;
  const int rb   = blockIdx.x;                   // rows [rb*16, rb*16+16)
  const int n0   = wave * 64;                    // this wave's 4 n-tiles

  // h0 = 0
  for (int i = tid; i < 16 * Hc; i += 256) hbuf[i] = f2bf(0.0f);

  // Preload xa slice for t=0: 8 rows x 512 f32 = 1024 x 16B chunks.
  #pragma unroll
  for (int k = 0; k < 4; ++k) {
    const int j    = tid + k * 256;              // 0..1023
    const int bloc = j >> 7;                     // batch row 0..7
    const int coff = (j & 127) * 4;              // col base
    async_g2lds_b128(xa + ((size_t)(rb * 8 + bloc) * Sc + 0) * Hc + coff,
                     &xabuf[0][bloc * Hc + coff]);
  }
  wait_async0();
  __syncthreads();

  const bf16* arow_lds = &hbuf[(size_t)l15 * Hc];
  int cur = 0;

  for (int t = 0; t < Sc; ++t) {
    // Kick off async prefetch of xa for step t+1 into the other buffer.
    if (t + 1 < Sc) {
      #pragma unroll
      for (int k = 0; k < 4; ++k) {
        const int j    = tid + k * 256;
        const int bloc = j >> 7;
        const int coff = (j & 127) * 4;
        async_g2lds_b128(xa + ((size_t)(rb * 8 + bloc) * Sc + (t + 1)) * Hc + coff,
                         &xabuf[cur ^ 1][bloc * Hc + coff]);
      }
    }

    // h * W_hh^T for this wave's 64 columns: 16 K-chunks x 4 N-tiles of WMMA.
    v8f acc[4] = {};
    for (int kc = 0; kc < Hc / 32; ++kc) {
      const int k0 = kc * 32 + half * 16;
      union { v16bf v; v8bf h[2]; } a;
      a.h[0] = *(const v8bf*)(arow_lds + k0);
      a.h[1] = *(const v8bf*)(arow_lds + k0 + 8);
      const int kb = kc * 32 + half * 16;
      #pragma unroll
      for (int nt = 0; nt < 4; ++nt) {
        const int g = n0 + nt * 16 + l15;        // B column = W_hh row
        v16bf b = *(const v16bf*)(whb + (size_t)g * Hc + kb);
        acc[nt] = __builtin_amdgcn_wmma_f32_16x16x32_bf16(
            false, a.v, false, b, (short)0, acc[nt], false, false);
      }
    }
    __syncthreads();                             // all reads of old h done

    #pragma unroll
    for (int nt = 0; nt < 4; ++nt) {
      const int col  = n0 + nt * 16 + l15;
      const float bias = bhh[col];
      #pragma unroll
      for (int r = 0; r < 8; ++r) {
        const int rloc = r + half * 8;
        const int grow = rb * 16 + rloc;         // global h row
        const int bidx = grow >> 1;              // b  (L = 2)
        const int lidx = grow & 1;               // layer index
        const float xav = xabuf[cur][(rloc >> 1) * Hc + col];
        const float v   = tanhf(acc[nt][r] + xav + bias);
        __builtin_nontemporal_store(
            v, &y[(((size_t)bidx * Sc + t) * Lc + lidx) * Hc + col]);
        hbuf[(size_t)rloc * Hc + col] = f2bf(v);
        if (t == Sc - 1)
          hlast[(size_t)grow * Hc + col] = v;
      }
    }

    wait_async0();                               // t+1 xa slice landed (per wave)
    __syncthreads();                             // h update + xa slice visible
    cur ^= 1;
  }
}

// ---------------------------------------------------------------------------
// Launch
// ---------------------------------------------------------------------------
extern "C" void kernel_launch(void* const* d_in, const int* in_sizes, int n_in,
                              void* d_out, int out_size, void* d_ws, size_t ws_size,
                              hipStream_t stream) {
  const float* x   = (const float*)d_in[0];   // [B,S,I]
  const float* Wih = (const float*)d_in[1];   // [H,I]
  const float* bih = (const float*)d_in[2];   // [H]
  const float* Whh = (const float*)d_in[3];   // [H,H]
  const float* bhh = (const float*)d_in[4];   // [H]
  // d_in[5] = num_layers (scalar), baked in as Lc=2.

  const size_t nX = (size_t)Bc * Sc * Ic;     // 16,777,216

  // Workspace layout (~97 MB): x_bf16 | W_ih_bf16 | W_hh_bf16 | xa_f32
  bf16*  xbf  = (bf16*)d_ws;
  bf16*  wihb = xbf  + nX;
  bf16*  whhb = wihb + (size_t)Hc * Ic;
  float* xa   = (float*)(whhb + (size_t)Hc * Hc);

  float* y     = (float*)d_out;
  float* hlast = y + (size_t)Bc * Sc * Lc * Hc;

  cvt_f32_bf16<<<4096, 256, 0, stream>>>(x,   xbf,  (int)nX);
  cvt_f32_bf16<<<256,  256, 0, stream>>>(Wih, wihb, Hc * Ic);
  cvt_f32_bf16<<<256,  256, 0, stream>>>(Whh, whhb, Hc * Hc);

  // 512 M-supertiles (64 rows) x 16 N-supertiles (32 cols) = 8192 wave-tiles
  gemm_xa_kernel<<<1024, 256, 0, stream>>>(xbf, wihb, bih, xa);

  // 8 row-blocks, each a persistent 16-row tile for all 512 steps
  rnn_scan_kernel<<<8, 256, 0, stream>>>(xa, whhb, bhh, y, hlast);
}
